// Attention_32993938768508
// MI455X (gfx1250) — matrix operand order repaired and checked by reading, not compiled, not run on hardware
//
#include <hip/hip_runtime.h>
#include <hip/hip_bf16.h>

typedef __attribute__((ext_vector_type(16))) _Float16 v16h;
typedef __attribute__((ext_vector_type(8)))  _Float16 v8h;
typedef __attribute__((ext_vector_type(4)))  _Float16 v4h;
typedef __attribute__((ext_vector_type(8)))  float    v8f;

#define DEV static __device__ __forceinline__

constexpr int Bsz = 2, Sq = 2048, Dm = 1024, NHh = 16, HDd = 64;

DEV v8f wmma16(v16h a, v16h b, v8f c) {
  // D = A(16x32 f16) * B(32x16 f16) + C(16x16 f32)
  return __builtin_amdgcn_wmma_f32_16x16x32_f16(false, a, false, b, (short)0, c,
                                                false, false);
}

// A fragment (16x32 f16). Lane L holds row M = L%16.
// Lanes 0-15: K in {kk..kk+7, kk+16..kk+23}; lanes 16-31: {kk+8..15, kk+24..31}.
DEV v16h load_afrag(const _Float16* __restrict__ rowptr, int kk, int lane) {
  const int off = kk + ((lane & 16) ? 8 : 0);
  v8h lo = *(const v8h*)(rowptr + off);
  v8h hi = *(const v8h*)(rowptr + off + 16);
  v16h a;
#pragma unroll
  for (int i = 0; i < 8; ++i) { a[i] = lo[i]; a[i + 8] = hi[i]; }
  return a;
}

// B fragment (32x16 f16). Lane L holds column N = L%16.
// Lanes 0-15: K in {kk..kk+7, kk+16..kk+23}; lanes 16-31: {kk+8..15, kk+24..31}.
DEV v16h load_bfrag(const _Float16* __restrict__ colptr, int kk, int lane) {
  const int off = kk + ((lane & 16) ? 8 : 0);
  v8h lo = *(const v8h*)(colptr + off);
  v8h hi = *(const v8h*)(colptr + off + 16);
  v16h b;
#pragma unroll
  for (int i = 0; i < 8; ++i) { b[i] = lo[i]; b[i + 8] = hi[i]; }
  return b;
}

// P fragment for PV: read f32 scores in A-frag layout, apply exp(x - m),
// accumulate per-lane partial row-sum, convert to f16.
DEV v16h load_pfrag(const float* __restrict__ rowptr, int j0, int lane, float m,
                    float& sum) {
  const int off = j0 + ((lane & 16) ? 8 : 0);
  const float4 a0 = *(const float4*)(rowptr + off);
  const float4 a1 = *(const float4*)(rowptr + off + 4);
  const float4 b0 = *(const float4*)(rowptr + off + 16);
  const float4 b1 = *(const float4*)(rowptr + off + 20);
  float x[16] = {a0.x, a0.y, a0.z, a0.w, a1.x, a1.y, a1.z, a1.w,
                 b0.x, b0.y, b0.z, b0.w, b1.x, b1.y, b1.z, b1.w};
  v16h p;
#pragma unroll
  for (int i = 0; i < 16; ++i) {
    const float e = __expf(x[i] - m);
    sum += e;
    p[i] = (_Float16)e;
  }
  return p;
}

__global__ void cvt_f32_to_f16(const float* __restrict__ src,
                               _Float16* __restrict__ dst, int n) {
  const int i = (blockIdx.x * blockDim.x + threadIdx.x) * 4;
  if (i < n) {
    const float4 v = *(const float4*)(src + i);
    v4h o;
    o[0] = (_Float16)v.x; o[1] = (_Float16)v.y;
    o[2] = (_Float16)v.z; o[3] = (_Float16)v.w;
    *(v4h*)(dst + i) = o;
  }
}

// C[M=4096, N=1024] = X[M,1024](f16) @ W[1024,1024]^T(f16) + bias
// MODE 0: write f16 at [B,NH,S,HD]   (Q, K)
// MODE 1: write f16 at [B,NH,HD,S]   (V transposed)
// MODE 2: write f32 at [M,N]         (output projection)
template <int MODE>
__global__ __launch_bounds__(256) void gemm_xwT(const _Float16* __restrict__ X,
                                                const _Float16* __restrict__ W,
                                                const float* __restrict__ bias,
                                                void* __restrict__ outp) {
  const int lane = threadIdx.x & 31;
  const int wave = threadIdx.x >> 5;
  const int tile = blockIdx.x * 8 + wave;      // 4096 tiles total
  const int NT = Dm / 64;                      // 16 n-supertiles
  const int m0 = (tile / NT) * 16;
  const int n0 = (tile % NT) * 64;
  const int ln = lane & 15;
  const int hw = (lane >> 4) & 1;

  const _Float16* arow = X + (size_t)(m0 + ln) * Dm;
  v8f zz = {};
  v8f acc[4];
#pragma unroll
  for (int t = 0; t < 4; ++t) acc[t] = zz;

  for (int kk = 0; kk < Dm; kk += 32) {
    const v16h a = load_afrag(arow, kk, lane);
#pragma unroll
    for (int t = 0; t < 4; ++t) {
      const _Float16* wrow = W + (size_t)(n0 + t * 16 + ln) * Dm;
      const v16h b = load_bfrag(wrow, kk, lane);
      acc[t] = wmma16(a, b, acc[t]);
    }
  }

  float bv[4];
#pragma unroll
  for (int t = 0; t < 4; ++t) bv[t] = bias[n0 + t * 16 + ln];

  const int bb = m0 >> 11;         // m / Sq
  const int s0 = m0 & (Sq - 1);
  const int hh = n0 >> 6;          // head index

#pragma unroll
  for (int t = 0; t < 4; ++t) {
#pragma unroll
    for (int r = 0; r < 8; ++r) {
      const int s = s0 + r + 8 * hw;
      const int d = t * 16 + ln;
      const float v = acc[t][r] + bv[t];
      if (MODE == 0) {
        _Float16* o = (_Float16*)outp;
        o[(((size_t)bb * NHh + hh) * Sq + s) * HDd + d] = (_Float16)v;
      } else if (MODE == 1) {
        _Float16* o = (_Float16*)outp;
        o[(((size_t)bb * NHh + hh) * HDd + d) * Sq + s] = (_Float16)v;
      } else {
        float* o = (float*)outp;
        o[(size_t)(m0 + r + 8 * hw) * Dm + n0 + t * 16 + ln] = v;
      }
    }
  }
}

// One wave per (b, h, 16-row q tile).
// Phase 1: scores = scale*QK^T + at + mask -> d_out, track row max.
// Phase 2: re-read scores, p = exp(x - m), PV WMMA, normalize by 1/sum.
__global__ __launch_bounds__(256) void attn_kernel(
    const _Float16* __restrict__ q16, const _Float16* __restrict__ k16,
    const _Float16* __restrict__ vT16, const float* __restrict__ at,
    const float* __restrict__ mask, float* __restrict__ scores,
    _Float16* __restrict__ ctx16) {
  __shared__ float smM[8][16];
  __shared__ float smS[8][32];
  const int lane = threadIdx.x & 31;
  const int wave = threadIdx.x >> 5;
  const int qt = blockIdx.x * 8 + wave;        // 4096 q tiles total
  const int bh = qt >> 7;                      // / (Sq/16)
  const int m0 = (qt & 127) << 4;
  const int b = bh >> 4;
  const int h = bh & 15;
  const int ln = lane & 15;
  const int hw = (lane >> 4) & 1;
  const float scale = 0.125f;                  // 1/sqrt(64)

  const _Float16* qrow = q16 + ((size_t)bh * Sq + m0 + ln) * HDd;
  const v16h qa0 = load_afrag(qrow, 0, lane);
  const v16h qa1 = load_afrag(qrow, 32, lane);

  float cmax[8];
#pragma unroll
  for (int r = 0; r < 8; ++r) cmax[r] = -3.0e38f;

  float* srow0 = scores + ((size_t)bh * Sq + m0) * Sq;
  const float* arow0 = at + ((size_t)bh * Sq + m0) * Sq;
  const float* mrow = mask + (size_t)b * Sq;

  for (int n0 = 0; n0 < Sq; n0 += 16) {
    const _Float16* krow = k16 + ((size_t)bh * Sq + n0 + ln) * HDd;
    const v16h kb0 = load_bfrag(krow, 0, lane);
    const v16h kb1 = load_bfrag(krow, 32, lane);
    v8f acc = {};
    acc = wmma16(qa0, kb0, acc);
    acc = wmma16(qa1, kb1, acc);
    const int col = n0 + ln;
    const float mv = mrow[col];
#pragma unroll
    for (int r = 0; r < 8; ++r) {
      const int row = r + 8 * hw;
      const float v = acc[r] * scale + arow0[(size_t)row * Sq + col] + mv;
      srow0[(size_t)row * Sq + col] = v;
      cmax[r] = fmaxf(cmax[r], v);
    }
  }

  // Row max: reduce across the 16 lanes of each half-wave.
#pragma unroll
  for (int r = 0; r < 8; ++r) {
    float v = cmax[r];
    v = fmaxf(v, __shfl_xor(v, 1));
    v = fmaxf(v, __shfl_xor(v, 2));
    v = fmaxf(v, __shfl_xor(v, 4));
    v = fmaxf(v, __shfl_xor(v, 8));
    cmax[r] = v;
  }
  if (ln == 0) {
#pragma unroll
    for (int r = 0; r < 8; ++r) smM[wave][hw * 8 + r] = cmax[r];
  }
  __syncthreads();
  const float mval = smM[wave][ln];            // A-frag row = ln

  v8f zz = {};
  v8f ctx[4];
#pragma unroll
  for (int t = 0; t < 4; ++t) ctx[t] = zz;
  float psum = 0.f;
  const float* prow = srow0 + (size_t)ln * Sq;

  for (int j0 = 0; j0 < Sq; j0 += 32) {
    const v16h pa = load_pfrag(prow, j0, lane, mval, psum);
#pragma unroll
    for (int t = 0; t < 4; ++t) {
      const _Float16* vrow = vT16 + ((size_t)bh * HDd + t * 16 + ln) * Sq;
      const v16h vb = load_bfrag(vrow, j0, lane);
      ctx[t] = wmma16(pa, vb, ctx[t]);
    }
  }

  smS[wave][lane] = psum;
  __syncthreads();
  float inv[8];
#pragma unroll
  for (int r = 0; r < 8; ++r) {
    const int row = r + 8 * hw;
    inv[r] = 1.0f / (smS[wave][row] + smS[wave][row + 16]);
  }
#pragma unroll
  for (int t = 0; t < 4; ++t) {
#pragma unroll
    for (int r = 0; r < 8; ++r) {
      const int row = r + 8 * hw;
      ctx16[((size_t)b * Sq + m0 + row) * Dm + h * HDd + t * 16 + ln] =
          (_Float16)(ctx[t][r] * inv[r]);
    }
  }
}

extern "C" void kernel_launch(void* const* d_in, const int* in_sizes, int n_in,
                              void* d_out, int out_size, void* d_ws,
                              size_t ws_size, hipStream_t stream) {
  (void)in_sizes; (void)n_in; (void)out_size; (void)ws_size;
  const float* hidden = (const float*)d_in[0];
  const float* amask  = (const float*)d_in[1];
  const float* at     = (const float*)d_in[2];
  const float* Wq = (const float*)d_in[3];
  const float* bq = (const float*)d_in[4];
  const float* Wk = (const float*)d_in[5];
  const float* bk = (const float*)d_in[6];
  const float* Wv = (const float*)d_in[7];
  const float* bv = (const float*)d_in[8];
  const float* Wo = (const float*)d_in[9];
  const float* bo = (const float*)d_in[10];

  float* out = (float*)d_out;                               // [B,S,D]
  float* scores = (float*)d_out + (size_t)Bsz * Sq * Dm;    // [B,NH,S,S]

  char* ws = (char*)d_ws;
  _Float16* h16  = (_Float16*)(ws);                         // 8 MB
  _Float16* wq16 = (_Float16*)(ws + ((size_t)8  << 20));    // 2 MB
  _Float16* wk16 = (_Float16*)(ws + ((size_t)10 << 20));
  _Float16* wv16 = (_Float16*)(ws + ((size_t)12 << 20));
  _Float16* wo16 = (_Float16*)(ws + ((size_t)14 << 20));
  _Float16* q16  = (_Float16*)(ws + ((size_t)16 << 20));    // 8 MB each
  _Float16* k16  = (_Float16*)(ws + ((size_t)24 << 20));
  _Float16* vT16 = (_Float16*)(ws + ((size_t)32 << 20));
  _Float16* c16  = (_Float16*)(ws + ((size_t)40 << 20));    // ends at 48 MB

  const int nH = Bsz * Sq * Dm;   // 4,194,304
  const int nW = Dm * Dm;         // 1,048,576
  cvt_f32_to_f16<<<(nH / 4 + 255) / 256, 256, 0, stream>>>(hidden, h16, nH);
  cvt_f32_to_f16<<<(nW / 4 + 255) / 256, 256, 0, stream>>>(Wq, wq16, nW);
  cvt_f32_to_f16<<<(nW / 4 + 255) / 256, 256, 0, stream>>>(Wk, wk16, nW);
  cvt_f32_to_f16<<<(nW / 4 + 255) / 256, 256, 0, stream>>>(Wv, wv16, nW);
  cvt_f32_to_f16<<<(nW / 4 + 255) / 256, 256, 0, stream>>>(Wo, wo16, nW);

  // 4096 (16x64) tiles, 8 waves per block -> 512 blocks
  gemm_xwT<0><<<512, 256, 0, stream>>>(h16, wq16, bq, (void*)q16);
  gemm_xwT<0><<<512, 256, 0, stream>>>(h16, wk16, bk, (void*)k16);
  gemm_xwT<1><<<512, 256, 0, stream>>>(h16, wv16, bv, (void*)vT16);

  // 4096 q-tiles, 8 waves per block -> 512 blocks
  attn_kernel<<<512, 256, 0, stream>>>(q16, k16, vT16, at, amask, scores, c16);

  gemm_xwT<2><<<512, 256, 0, stream>>>(c16, wo16, bo, (void*)out);
}
